// VMUNetEncoder_69449621176624
// MI455X (gfx1250) — compile-verified, matrix operand order checked by reading, not run on hardware
//
#include <hip/hip_runtime.h>
#include <hip/hip_bf16.h>
#include <math.h>

typedef __attribute__((ext_vector_type(16))) __bf16 v16bf;
typedef __attribute__((ext_vector_type(8)))  __bf16 v8bf;
typedef __attribute__((ext_vector_type(8)))  float  v8f;
typedef unsigned int uint32x4 __attribute__((ext_vector_type(4)));
typedef int          int32x4  __attribute__((ext_vector_type(4)));
typedef int          int32x8  __attribute__((ext_vector_type(8)));

#define LDSP 40   // LDS row stride in bf16 elements (80 B, multiple of 16 B)

__device__ __forceinline__ float silu_f(float x) { return x / (1.f + __expf(-x)); }

// ---------------------------------------------------------------------------
// Stem: conv7x7 stride2 pad3 + BatchNorm(eval) + exact GELU. Token-major out.
// ---------------------------------------------------------------------------
__global__ void k_stem(const float* __restrict__ x, const float* __restrict__ w,
                       const float* __restrict__ bias,
                       const float* __restrict__ bn_g, const float* __restrict__ bn_b,
                       const float* __restrict__ bn_m, const float* __restrict__ bn_v,
                       float* __restrict__ out, int Bn, int Cin, int Hin, int Win,
                       int Cout, int Hout, int Wout)
{
    long idx = (long)blockIdx.x * blockDim.x + threadIdx.x;
    long total = (long)Bn * Hout * Wout * Cout;
    if (idx >= total) return;
    int co = (int)(idx % Cout);
    long t = idx / Cout;
    int ww = (int)(t % Wout); t /= Wout;
    int hh = (int)(t % Hout); int b = (int)(t / Hout);

    float acc = bias[co];
    for (int ci = 0; ci < Cin; ++ci) {
        const float* xp = x + ((long)(b * Cin + ci) * Hin) * Win;
        const float* wp = w + (long)(co * Cin + ci) * 49;
#pragma unroll
        for (int kh = 0; kh < 7; ++kh) {
            int ih = hh * 2 - 3 + kh;
            if (ih < 0 || ih >= Hin) continue;
#pragma unroll
            for (int kw = 0; kw < 7; ++kw) {
                int iw = ww * 2 - 3 + kw;
                if (iw < 0 || iw >= Win) continue;
                acc += wp[kh * 7 + kw] * xp[(long)ih * Win + iw];
            }
        }
    }
    acc = (acc - bn_m[co]) * rsqrtf(bn_v[co] + 1e-5f) * bn_g[co] + bn_b[co];
    acc = 0.5f * acc * (1.f + erff(acc * 0.70710678118654752f));
    out[idx] = acc;
}

// ---------------------------------------------------------------------------
// LayerNorm over last dim D; one wave32 per token.
// ---------------------------------------------------------------------------
__global__ void k_layernorm(const float* __restrict__ x, const float* __restrict__ g,
                            const float* __restrict__ b, float* __restrict__ y,
                            int M, int D)
{
    int wave = threadIdx.x >> 5;
    int lane = threadIdx.x & 31;
    int t = blockIdx.x * (blockDim.x >> 5) + wave;
    if (t >= M) return;
    const float* xr = x + (size_t)t * D;
    float s = 0.f, sq = 0.f;
    for (int c = lane; c < D; c += 32) { float v = xr[c]; s += v; sq += v * v; }
#pragma unroll
    for (int off = 16; off > 0; off >>= 1) {
        s  += __shfl_xor(s,  off, 32);
        sq += __shfl_xor(sq, off, 32);
    }
    float mu = s / (float)D;
    float var = sq / (float)D - mu * mu;
    float rstd = rsqrtf(var + 1e-5f);
    float* yr = y + (size_t)t * D;
    for (int c = lane; c < D; c += 32) yr[c] = (xr[c] - mu) * rstd * g[c] + b[c];
}

// ---------------------------------------------------------------------------
// Weight pre-pass: f32 (N x K, row stride ldw) -> bf16 (Np x Kp), zero padded.
// ---------------------------------------------------------------------------
__global__ void k_cvt_w(const float* __restrict__ W, int ldw, int N, int K,
                        int Kp, long total, void* __restrict__ outv)
{
    __bf16* out = (__bf16*)outv;
    long idx = (long)blockIdx.x * blockDim.x + threadIdx.x;
    if (idx >= total) return;
    int k = (int)(idx % Kp);
    int n = (int)(idx / Kp);
    float v = 0.f;
    if (n < N && k < K) v = W[(size_t)n * ldw + k];
    out[idx] = (__bf16)v;
}

// ---------------------------------------------------------------------------
// GEMM: C[M,N] = A[M,K](f32) * Wb[N,K](bf16,padded)^T (+bias)(act)(+resid)
// 256 threads = 8 waves; workgroup tile 64(M) x 64(N); K stepped by 32.
// A tile: branch-free clamped float4 loads -> bf16 -> LDS (b128 stores).
// W tile: Tensor Data Mover (tensor_load_to_lds) with LDS padding -> 40-elem
//         row stride, so all fragment reads are ds_load_b128.
// Two v_wmma_f32_16x16x32_bf16 per wave per K-step.
// act: 0 = none, 1 = softplus.
// Call-site requirements (all satisfied here): lda multiple of 4; reading 32
// floats from a row starting at any 32-aligned k0 < K stays inside the row.
// ---------------------------------------------------------------------------
__global__ __launch_bounds__(256) void k_gemm(
        const float* __restrict__ A, int lda,
        const void* __restrict__ Wbv, int Kp,
        const float* __restrict__ bias,
        const float* __restrict__ resid, int ldr,
        float* __restrict__ C, int ldc,
        int M, int N, int K, int act)
{
    __shared__ __bf16 As[64 * LDSP];
    __shared__ __bf16 Bs[64 * LDSP];

    int tid  = threadIdx.x;
    int lane = tid & 31;
    int wave = tid >> 5;
    int wm = wave >> 1;          // 0..3 : 16-row tile
    int wq = wave & 1;           // 0..1 : 32-col super-tile
    int lr = lane & 15;
    int hi = lane >> 4;

    int blockM = blockIdx.y * 64;
    int blockN = blockIdx.x * 64;

    // A tile mapping: row = tid>>2 (0..63), 8 consecutive k per thread.
    int arow  = tid >> 2;
    int akoff = (tid & 3) * 8;
    int am = blockM + arow; if (am > M - 1) am = M - 1;   // clamp: branch-free
    const float* aptr = A + (size_t)am * lda + akoff;

    // ---- TDM descriptor (uniform across workgroup) ----
    unsigned long long gbase = (unsigned long long)(uintptr_t)Wbv
                             + ((unsigned long long)blockN * (unsigned)Kp) * 2ull;
    unsigned int lds_b = (unsigned int)(uintptr_t)(&Bs[0]);
    // group1 word0: data_size=1 (2B) | pad_enable | pad_interval=3 (16 dw)
    //               | pad_amount=3 (4 dw)  -> LDS row stride = 40 bf16
    unsigned int g1w0 = (1u << 16) | (1u << 20) | (3u << 22) | (3u << 25);
    unsigned int kp16 = (unsigned int)Kp & 0xFFFFu;

    v8f acc0 = {}, acc1 = {};

    for (int k0 = 0; k0 < K; k0 += 32) {
        // -- issue weight-tile DMA (one wave; TDM ignores EXEC) --
        if (wave == 0) {
            unsigned long long ga = gbase + (unsigned long long)k0 * 2ull;
            uint32x4 g0;
            g0[0] = 1u;                                   // count=1, no gather
            g0[1] = lds_b;                                // LDS address
            g0[2] = (unsigned int)(ga & 0xFFFFFFFFull);
            g0[3] = (unsigned int)((ga >> 32) & 0x01FFFFFFull) | (2u << 30);
            int32x8 g1;
            g1[0] = (int)g1w0;
            g1[1] = (int)(kp16 << 16);                    // tensor_dim0 lo16
            g1[2] = (int)(64u << 16);                     // tensor_dim1=Np rows >= 64*gridN; use tile-local 64
            g1[3] = (int)(32u << 16);                     // tile_dim0 = 32
            g1[4] = (int)64;                              // tile_dim1 = 64 rows
            g1[5] = (int)(unsigned)Kp;                    // tensor_dim0_stride
            g1[6] = 0;
            g1[7] = 0;
            int32x4 z4 = {};
#if defined(__clang_major__) && (__clang_major__ >= 23)
            int32x8 z8 = {};
            __builtin_amdgcn_tensor_load_to_lds(g0, g1, z4, z4, z8, 0);
#else
            __builtin_amdgcn_tensor_load_to_lds(g0, g1, z4, z4, 0);
#endif
        }

        // -- A tile: two b128 loads, convert, one b128 LDS store --
        float4 f0 = *(const float4*)(aptr + k0);
        float4 f1 = *(const float4*)(aptr + k0 + 4);
        __builtin_prefetch(aptr + k0 + 32, 0, 1);         // global_prefetch_b8
        v8bf a8;
        a8[0] = (__bf16)f0.x; a8[1] = (__bf16)f0.y;
        a8[2] = (__bf16)f0.z; a8[3] = (__bf16)f0.w;
        a8[4] = (__bf16)f1.x; a8[5] = (__bf16)f1.y;
        a8[6] = (__bf16)f1.z; a8[7] = (__bf16)f1.w;
        *(v8bf*)&As[arow * LDSP + akoff] = a8;

        if (wave == 0) __builtin_amdgcn_s_wait_tensorcnt(0);
        __syncthreads();

        // -- fragments: two ds_load_b128 per operand, concat --
        const __bf16* ar = &As[(wm * 16 + lr) * LDSP + hi * 8];
        v8bf alo = *(const v8bf*)ar;
        v8bf ahi = *(const v8bf*)(ar + 16);
        v16bf af = __builtin_shufflevector(alo, ahi,
                    0,1,2,3,4,5,6,7,8,9,10,11,12,13,14,15);

        int nb = wq * 32;
        const __bf16* br0 = &Bs[(nb + lr) * LDSP + hi * 8];
        const __bf16* br1 = &Bs[(nb + 16 + lr) * LDSP + hi * 8];
        v8bf b0l = *(const v8bf*)br0;
        v8bf b0h = *(const v8bf*)(br0 + 16);
        v8bf b1l = *(const v8bf*)br1;
        v8bf b1h = *(const v8bf*)(br1 + 16);
        v16bf bf0 = __builtin_shufflevector(b0l, b0h,
                    0,1,2,3,4,5,6,7,8,9,10,11,12,13,14,15);
        v16bf bf1 = __builtin_shufflevector(b1l, b1h,
                    0,1,2,3,4,5,6,7,8,9,10,11,12,13,14,15);

        acc0 = __builtin_amdgcn_wmma_f32_16x16x32_bf16(
                   false, af, false, bf0, (short)0, acc0, false, false);
        acc1 = __builtin_amdgcn_wmma_f32_16x16x32_bf16(
                   false, af, false, bf1, (short)0, acc1, false, false);
        __syncthreads();
    }

    // -- epilogue --
#pragma unroll
    for (int r = 0; r < 8; ++r) {
        int row  = blockM + wm * 16 + hi * 8 + r;
        int col0 = blockN + wq * 32 + lr;
        int col1 = col0 + 16;
        if (row < M) {
            if (col0 < N) {
                float v = acc0[r];
                if (bias)     v += bias[col0];
                if (act == 1) v = (v > 20.f) ? v : log1pf(__expf(v));
                if (resid)    v += resid[(size_t)row * ldr + col0];
                C[(size_t)row * ldc + col0] = v;
            }
            if (col1 < N) {
                float v = acc1[r];
                if (bias)     v += bias[col1];
                if (act == 1) v = (v > 20.f) ? v : log1pf(__expf(v));
                if (resid)    v += resid[(size_t)row * ldr + col1];
                C[(size_t)row * ldc + col1] = v;
            }
        }
    }
}

// ---------------------------------------------------------------------------
// Causal depthwise conv1d (K=4) + bias + SiLU.
// ---------------------------------------------------------------------------
__global__ void k_dwconv_silu(const float* __restrict__ xz, const float* __restrict__ w,
                              const float* __restrict__ bias, float* __restrict__ xc,
                              int B, int L, int di)
{
    long idx = (long)blockIdx.x * blockDim.x + threadIdx.x;
    long total = (long)B * L * di;
    if (idx >= total) return;
    int c = (int)(idx % di);
    long t = idx / di;
    int l = (int)(t % L);
    int b = (int)(t / L);
    int ld = 2 * di;
    const float* base = xz + ((size_t)b * L) * ld + c;
    float acc = bias[c];
#pragma unroll
    for (int j = 0; j < 4; ++j) {
        int ll = l - 3 + j;
        if (ll >= 0) acc += w[c * 4 + j] * base[(size_t)ll * ld];
    }
    xc[idx] = silu_f(acc);
}

// ---------------------------------------------------------------------------
// Selective scan, one lane per (batch, channel), fused D-skip + z-gate.
// ---------------------------------------------------------------------------
__global__ void k_scan(const float* __restrict__ dt, const float* __restrict__ dbl, int dtr,
                       const float* __restrict__ xc, const float* __restrict__ xz,
                       const float* __restrict__ A_log, const float* __restrict__ Dp,
                       float* __restrict__ yg, int B, int L, int di)
{
    int g = blockIdx.x * blockDim.x + threadIdx.x;
    if (g >= B * di) return;
    int b = g / di, c = g % di;
    float a[16], h[16];
#pragma unroll
    for (int n = 0; n < 16; ++n) { a[n] = -__expf(A_log[c * 16 + n]); h[n] = 0.f; }
    float Dc = Dp[c];
    int ld2 = 2 * di;
    for (int l = 0; l < L; ++l) {
        size_t m = (size_t)b * L + l;
        float dtv = dt[m * di + c];
        float xv  = xc[m * di + c];
        float zv  = xz[m * ld2 + di + c];
        const float* Bp = dbl + m * 48 + dtr;
        const float* Cp = Bp + 16;
        float y = 0.f;
#pragma unroll
        for (int n = 0; n < 16; ++n) {
            h[n] = __expf(dtv * a[n]) * h[n] + dtv * Bp[n] * xv;
            y += h[n] * Cp[n];
        }
        yg[m * di + c] = (y + xv * Dc) * silu_f(zv);
    }
}

// ---------------------------------------------------------------------------
// Token-major (B, HW, C) -> NCHW feature emit.
// ---------------------------------------------------------------------------
__global__ void k_tok2nchw(const float* __restrict__ in, float* __restrict__ out,
                           int B, int HW, int C)
{
    long idx = (long)blockIdx.x * blockDim.x + threadIdx.x;
    long total = (long)B * HW * C;
    if (idx >= total) return;
    int c = (int)(idx % C);
    long t = idx / C;
    int p = (int)(t % HW);
    int b = (int)(t / HW);
    out[((size_t)b * C + c) * HW + p] = in[idx];
}

// ---------------------------------------------------------------------------
// Downsample conv 2x2 stride2, token-major in -> token-major out.
// ---------------------------------------------------------------------------
__global__ void k_down(const float* __restrict__ in, const float* __restrict__ w,
                       const float* __restrict__ bias, float* __restrict__ out,
                       int B, int H, int Wd, int Cin, int Cout)
{
    int Ho = H >> 1, Wo = Wd >> 1;
    long idx = (long)blockIdx.x * blockDim.x + threadIdx.x;
    long total = (long)B * Ho * Wo * Cout;
    if (idx >= total) return;
    int co = (int)(idx % Cout);
    long t = idx / Cout;
    int ow = (int)(t % Wo); t /= Wo;
    int oh = (int)(t % Ho); int b = (int)(t / Ho);
    float acc = bias[co];
    for (int ci = 0; ci < Cin; ++ci) {
#pragma unroll
        for (int kh = 0; kh < 2; ++kh)
#pragma unroll
            for (int kw = 0; kw < 2; ++kw)
                acc += w[((co * Cin + ci) * 2 + kh) * 2 + kw] *
                       in[(((size_t)b * H + (oh * 2 + kh)) * Wd + (ow * 2 + kw)) * Cin + ci];
    }
    out[idx] = acc;
}

// ---------------------------------------------------------------------------
// Host orchestration.
// ---------------------------------------------------------------------------
static inline void gemm_full(const float* A, int lda, const float* W, int ldw,
                             const float* bias, const float* resid, int ldr,
                             float* C, int ldc, int M, int N, int K, int act,
                             void* WBF, hipStream_t s)
{
    int Kp = (K + 31) & ~31;
    int Np = (N + 63) & ~63;
    long tot = (long)Np * Kp;
    k_cvt_w<<<(unsigned)((tot + 255) / 256), 256, 0, s>>>(W, ldw, N, K, Kp, tot, WBF);
    dim3 grid((N + 63) / 64, (M + 63) / 64);
    k_gemm<<<grid, 256, 0, s>>>(A, lda, WBF, Kp, bias, resid, ldr, C, ldc, M, N, K, act);
}

extern "C" void kernel_launch(void* const* d_in, const int* in_sizes, int n_in,
                              void* d_out, int out_size, void* d_ws, size_t ws_size,
                              hipStream_t stream)
{
    (void)in_sizes; (void)n_in; (void)out_size; (void)ws_size;
    const int Bn = 8;
    const int dims[4] = {32, 64, 128, 256};
    const int Hs[4]   = {128, 64, 32, 16};

    auto P = [&](int i) { return (const float*)d_in[i]; };
    // d_in layout (dict insertion order, recursive):
    // 0: x | 1..6: stem_w, stem_b, bn_g, bn_b, bn_m, bn_v
    // 7..94: stages[4][2] x 11: ln_g ln_b in_proj conv_w conv_b x_proj
    //                           dt_w dt_b A_log D out_proj
    // 95..100: downs[3] {w, b}

    float* ws   = (float*)d_ws;
    float* bufA = ws;                    //  4,194,304 floats (max M*D)
    float* bufB = ws + 4194304;          //  4,194,304
    float* XZ   = ws + 8388608;          // 16,777,216 (max M*2di)
    float* XC   = ws + 25165824;         //  8,388,608 (max M*di)
    float* DBL  = ws + 33554432;         //  6,291,456 (M*48)
    float* DT   = ws + 39845888;         //  8,388,608
    float* YG   = ws + 48234496;         //  8,388,608
    void*  WBF  = (void*)(ws + 56623104);// 262,144 bf16 (131,072 floats)

    float* X    = bufA;
    float* Xalt = bufB;

    // ---- stem ----
    {
        long total = (long)Bn * 128 * 128 * 32;
        k_stem<<<(unsigned)((total + 255) / 256), 256, 0, stream>>>(
            P(0), P(1), P(2), P(3), P(4), P(5), P(6),
            X, Bn, 3, 256, 256, 32, 128, 128);
    }

    size_t out_off = 0;
    for (int si = 0; si < 4; ++si) {
        int d  = dims[si];
        int H  = Hs[si];
        int L  = H * H;
        int M  = Bn * L;
        int di = 2 * d;
        int dtr = (d + 15) / 16;

        for (int j = 0; j < 2; ++j) {
            int pb = 7 + (si * 2 + j) * 11;
            const float* ln_g   = P(pb + 0);
            const float* ln_b   = P(pb + 1);
            const float* inproj = P(pb + 2);
            const float* conv_w = P(pb + 3);
            const float* conv_b = P(pb + 4);
            const float* xproj  = P(pb + 5);
            const float* dt_w   = P(pb + 6);
            const float* dt_b   = P(pb + 7);
            const float* A_log  = P(pb + 8);
            const float* Dp     = P(pb + 9);
            const float* oproj  = P(pb + 10);

            // 1. LayerNorm -> Xalt (M x d)
            k_layernorm<<<(M + 7) / 8, 256, 0, stream>>>(X, ln_g, ln_b, Xalt, M, d);
            // 2. in_proj: XZ (M x 2di)
            gemm_full(Xalt, d, inproj, d, nullptr, nullptr, 0,
                      XZ, 2 * di, M, 2 * di, d, 0, WBF, stream);
            // 3. depthwise causal conv + SiLU -> XC (M x di)
            {
                long total = (long)M * di;
                k_dwconv_silu<<<(unsigned)((total + 255) / 256), 256, 0, stream>>>(
                    XZ, conv_w, conv_b, XC, Bn, L, di);
            }
            // 4. x_proj: DBL (M x 48, valid cols dtr+32)
            gemm_full(XC, di, xproj, di, nullptr, nullptr, 0,
                      DBL, 48, M, dtr + 32, di, 0, WBF, stream);
            // 5. dt: DT (M x di) = softplus(DBL[:, :dtr] * dt_w^T + dt_b)
            gemm_full(DBL, 48, dt_w, dtr, dt_b, nullptr, 0,
                      DT, di, M, di, dtr, 1, WBF, stream);
            // 6. selective scan + gate -> YG (M x di)
            {
                int threads = Bn * di;
                k_scan<<<(threads + 255) / 256, 256, 0, stream>>>(
                    DT, DBL, dtr, XC, XZ, A_log, Dp, YG, Bn, L, di);
            }
            // 7. out_proj + residual -> X (M x d)
            gemm_full(YG, di, oproj, di, nullptr, X, d,
                      X, d, M, d, di, 0, WBF, stream);
        }

        // emit feature map (NCHW)
        {
            long total = (long)M * d;
            k_tok2nchw<<<(unsigned)((total + 255) / 256), 256, 0, stream>>>(
                X, (float*)d_out + out_off, Bn, L, d);
            out_off += (size_t)M * d;
        }

        // downsample
        if (si < 3) {
            const float* dw = P(95 + si * 2);
            const float* db = P(96 + si * 2);
            int Cout = dims[si + 1];
            long total = (long)Bn * (H / 2) * (H / 2) * Cout;
            k_down<<<(unsigned)((total + 255) / 256), 256, 0, stream>>>(
                X, dw, db, Xalt, Bn, H, H, d, Cout);
            float* tmp = X; X = Xalt; Xalt = tmp;
        }
    }
}
